// MoeExperts_35759897706715
// MI455X (gfx1250) — matrix operand order repaired and checked by reading, not compiled, not run on hardware
//
#include <hip/hip_runtime.h>
#include <hip/hip_bf16.h>

// ---------------------------------------------------------------------------
// MoE combine (T=16384, H=1024, F=4096, E=8, top-k K=2), fp32 in/out.
// Strategy:
//   1) convert hidden + weights to bf16 (weights transposed to [N][K]) in ws
//   2) route tokens: per (slot j, expert e) compacted token/gate lists
//   3) two deterministic passes (j=0 writes out, j=1 accumulates):
//      fused FFN per 32-token tile using v_wmma_f32_16x16x32_bf16,
//      h-chunk staged in LDS, out tile held in 128 accumulator VGPRs/wave.
//      __launch_bounds__(256, 4) caps VGPRs under the 256 direct window
//      (no vgpr-msb switching) and raises resident waves per SIMD.
// ---------------------------------------------------------------------------

namespace {
constexpr int T_TOK  = 16384;
constexpr int H_DIM  = 1024;
constexpr int F_DIM  = 4096;
constexpr int E_NUM  = 8;
constexpr int K_SLOT = 2;
constexpr int MTILE  = 32;   // tokens per workgroup
constexpr int FCHUNK = 128;  // F columns per fused chunk (8 waves x 16)
}

typedef __bf16 bf16;
typedef bf16  bf16x4 __attribute__((ext_vector_type(4)));
typedef bf16  bf16x8 __attribute__((ext_vector_type(8)));
typedef bf16  v16bf  __attribute__((ext_vector_type(16)));
typedef float v8f    __attribute__((ext_vector_type(8)));

// A fragment (16x32 bf16): per lane two contiguous 8-elem runs at p and p+16.
__device__ __forceinline__ v16bf lda_frag(const bf16* p) {
  bf16x8 lo = *reinterpret_cast<const bf16x8*>(p);
  bf16x8 hi = *reinterpret_cast<const bf16x8*>(p + 16);
  return __builtin_shufflevector(lo, hi, 0,1,2,3,4,5,6,7,8,9,10,11,12,13,14,15);
}
// B fragment (32x16 bf16, weights stored [N][K]): two contiguous 8-elem runs at p and p+16.
__device__ __forceinline__ v16bf ldb_frag(const bf16* p) {
  bf16x8 lo = *reinterpret_cast<const bf16x8*>(p);
  bf16x8 hi = *reinterpret_cast<const bf16x8*>(p + 16);
  return __builtin_shufflevector(lo, hi, 0,1,2,3,4,5,6,7,8,9,10,11,12,13,14,15);
}

__device__ __forceinline__ float gelu_exact(float x) {
  return 0.5f * x * (1.0f + erff(x * 0.7071067811865476f));
}

#define WMMA_BF16(a, b, c) \
  __builtin_amdgcn_wmma_f32_16x16x32_bf16(false, (a), false, (b), (short)0, (c), false, false)

// ---------------------------------------------------------------------------
__global__ void zero_counts_kernel(int* counts) {
  if (threadIdx.x < K_SLOT * E_NUM) counts[threadIdx.x] = 0;
}

__global__ void route_kernel(const int* __restrict__ indices,
                             const float* __restrict__ probs,
                             int* __restrict__ counts,
                             int* __restrict__ lists,
                             float* __restrict__ gates) {
  int t = blockIdx.x * blockDim.x + threadIdx.x;
  if (t >= T_TOK) return;
#pragma unroll
  for (int j = 0; j < K_SLOT; ++j) {
    int   e = indices[t * K_SLOT + j];
    float p = probs[t * K_SLOT + j];
    int slot = j * E_NUM + e;
    int pos  = atomicAdd(&counts[slot], 1);
    lists[(size_t)slot * T_TOK + pos] = t;
    gates[(size_t)slot * T_TOK + pos] = p;
  }
}

__global__ void cvt_x_kernel(const float* __restrict__ x, bf16* __restrict__ xb, int n4) {
  int i = blockIdx.x * blockDim.x + threadIdx.x;
  if (i >= n4) return;
  float4 v = reinterpret_cast<const float4*>(x)[i];
  bf16x4 o = { (bf16)v.x, (bf16)v.y, (bf16)v.z, (bf16)v.w };
  reinterpret_cast<bf16x4*>(xb)[i] = o;
}

// src: [E][R][C] fp32  ->  dst: [E][C][R] bf16   (tiled transpose via LDS)
__global__ __launch_bounds__(256)
void transpose_cvt_kernel(const float* __restrict__ src, bf16* __restrict__ dst,
                          int R, int C) {
  __shared__ float tile[32][33];
  int e  = blockIdx.z;
  const float* s = src + (size_t)e * R * C;
  bf16*        d = dst + (size_t)e * R * C;
  int c0 = blockIdx.x * 32, r0 = blockIdx.y * 32;
  int tx = threadIdx.x, ty = threadIdx.y;
#pragma unroll
  for (int i = 0; i < 32; i += 8)
    tile[ty + i][tx] = s[(size_t)(r0 + ty + i) * C + (c0 + tx)];
  __syncthreads();
#pragma unroll
  for (int i = 0; i < 32; i += 8)
    d[(size_t)(c0 + ty + i) * R + (r0 + tx)] = (bf16)tile[tx][ty + i];
}

// ---------------------------------------------------------------------------
// Fused expert FFN over a compacted 32-token tile.
// Block = 256 threads (8 waves). Wave w owns F columns [fc + w*16) in GEMM1 and
// H columns [w*128, (w+1)*128) in GEMM2 (16 C-tiles = 128 accumulator VGPRs).
// min 4 waves/EU caps register allocation below the 256-VGPR direct window.
__global__ __launch_bounds__(256, 4)
void moe_expert_fused(const bf16* __restrict__ Xb, const bf16* __restrict__ W1b,
                      const bf16* __restrict__ W2b, const float* __restrict__ b1,
                      const float* __restrict__ b2, const int* __restrict__ counts,
                      const int* __restrict__ lists, const float* __restrict__ gates,
                      float* __restrict__ out, int pass, int accum) {
  const int e     = blockIdx.y;
  const int slot  = pass * E_NUM + e;
  const int cnt   = counts[slot];
  const int tile0 = blockIdx.x * MTILE;
  if (tile0 >= cnt) return;                   // uniform early-out
  const int rowsValid = min(MTILE, cnt - tile0);

  __shared__ __align__(16) bf16 hs[MTILE][FCHUNK + 8];  // +8 keeps 16B row align
  __shared__ int   rowTok[MTILE];
  __shared__ float rowGate[MTILE];

  const int tid = threadIdx.x;
  if (tid < MTILE) {
    int idx = tile0 + tid;
    bool valid = idx < cnt;
    rowTok[tid]  = lists[(size_t)slot * T_TOK + (valid ? idx : tile0)];
    rowGate[tid] = valid ? gates[(size_t)slot * T_TOK + idx] : 0.0f;
  }
  __syncthreads();

  const int wave = tid >> 5;
  const int lane = tid & 31;
  const int l16  = lane & 15;
  const int lh   = lane >> 4;

  // Gathered token rows for the two M-tiles (A operands, bf16 from global/L2).
  const bf16* a0 = Xb + (size_t)rowTok[l16]      * H_DIM + lh * 8;
  const bf16* a1 = Xb + (size_t)rowTok[16 + l16] * H_DIM + lh * 8;

  // GEMM2 B base pointers: W2b is [E][H][F] = [N][K], contiguous in K.
  const bf16* bp2[8];
#pragma unroll
  for (int nt = 0; nt < 8; ++nt) {
    int n2 = wave * 128 + nt * 16 + l16;
    bp2[nt] = W2b + ((size_t)e * H_DIM + n2) * F_DIM + lh * 8;
  }

  v8f zero = {};
  v8f acc[16];
#pragma unroll
  for (int i = 0; i < 16; ++i) acc[i] = zero;

  for (int fc = 0; fc < F_DIM; fc += FCHUNK) {
    // ---- GEMM1: h[:, fc + wave*16 .. +15] = x @ W1 (W1b is [E][F][H] = [N][K])
    // 1-stage software pipeline: fragments for step k+32 are issued before the
    // WMMAs for step k, so each wait covers a full iteration of latency.
    const int n1 = fc + wave * 16 + l16;
    const bf16* bp1 = W1b + ((size_t)e * F_DIM + n1) * H_DIM + lh * 8;
    v8f c0 = zero, c1 = zero;
    v16bf av0 = lda_frag(a0);
    v16bf av1 = lda_frag(a1);
    v16bf bv  = ldb_frag(bp1);
#pragma unroll 4
    for (int k = 32; k < H_DIM; k += 32) {
      v16bf nav0 = lda_frag(a0 + k);
      v16bf nav1 = lda_frag(a1 + k);
      v16bf nbv  = ldb_frag(bp1 + k);
      c0 = WMMA_BF16(av0, bv, c0);
      c1 = WMMA_BF16(av1, bv, c1);
      av0 = nav0; av1 = nav1; bv = nbv;
    }
    c0 = WMMA_BF16(av0, bv, c0);
    c1 = WMMA_BF16(av1, bv, c1);

    // bias + exact GELU, stage bf16 h-chunk into LDS
    const float b1v = b1[(size_t)e * F_DIM + n1];
    const int   col = wave * 16 + l16;
#pragma unroll
    for (int v = 0; v < 8; ++v) {           // C layout: VGPR v -> M = v + 8*lh
      hs[v + 8 * lh][col]      = (bf16)gelu_exact(c0[v] + b1v);
      hs[16 + v + 8 * lh][col] = (bf16)gelu_exact(c1[v] + b1v);
    }
    __syncthreads();

    // ---- GEMM2: acc += h_chunk @ W2[fc:fc+128, :]
    // B fragments batched in groups of 4: 4-deep load clauses with partial
    // LOADcnt waits under the capped register budget.
#pragma unroll
    for (int kk = 0; kk < FCHUNK; kk += 32) {
      v16bf h0 = lda_frag(&hs[l16][kk + lh * 8]);
      v16bf h1 = lda_frag(&hs[16 + l16][kk + lh * 8]);
#pragma unroll
      for (int g = 0; g < 2; ++g) {
        v16bf bvs[4];
#pragma unroll
        for (int i = 0; i < 4; ++i) bvs[i] = ldb_frag(bp2[g * 4 + i] + fc + kk);
#pragma unroll
        for (int i = 0; i < 4; ++i) {
          int nt = g * 4 + i;
          acc[nt]     = WMMA_BF16(h0, bvs[i], acc[nt]);
          acc[8 + nt] = WMMA_BF16(h1, bvs[i], acc[8 + nt]);
        }
      }
    }
    __syncthreads();
  }

  // ---- epilogue: out[tok, n] (pass0 write / pass1 accumulate) with gate + b2
  int toks[16]; float gts[16]; int rws[16];
#pragma unroll
  for (int mt = 0; mt < 2; ++mt)
#pragma unroll
    for (int v = 0; v < 8; ++v) {
      int row = mt * 16 + 8 * lh + v;
      toks[mt * 8 + v] = rowTok[row];
      gts[mt * 8 + v]  = rowGate[row];
      rws[mt * 8 + v]  = row;
    }
#pragma unroll
  for (int nt = 0; nt < 8; ++nt) {
    int   n2  = wave * 128 + nt * 16 + l16;
    float b2v = b2[(size_t)e * H_DIM + n2];
#pragma unroll
    for (int mt = 0; mt < 2; ++mt) {
      v8f c = acc[mt * 8 + nt];
#pragma unroll
      for (int v = 0; v < 8; ++v) {
        int q = mt * 8 + v;
        if (rws[q] < rowsValid) {
          float  val = gts[q] * (c[v] + b2v);
          float* p   = out + (size_t)toks[q] * H_DIM + n2;
          if (accum) *p += val; else *p = val;
        }
      }
    }
  }
}

// ---------------------------------------------------------------------------
extern "C" void kernel_launch(void* const* d_in, const int* in_sizes, int n_in,
                              void* d_out, int out_size, void* d_ws, size_t ws_size,
                              hipStream_t stream) {
  const float* hidden  = (const float*)d_in[0];
  const float* probs   = (const float*)d_in[1];
  const int*   indices = (const int*)d_in[2];
  const float* W1      = (const float*)d_in[3];
  const float* b1      = (const float*)d_in[4];
  const float* W2      = (const float*)d_in[5];
  const float* b2      = (const float*)d_in[6];
  float*       out     = (float*)d_out;
  char*        ws      = (char*)d_ws;

  const size_t szXb  = (size_t)T_TOK * H_DIM * sizeof(bf16);          // 32 MiB
  const size_t szW   = (size_t)E_NUM * H_DIM * F_DIM * sizeof(bf16);  // 64 MiB each
  size_t offXb   = 0;
  size_t offW1   = offXb + szXb;
  size_t offW2   = offW1 + szW;
  size_t offCnt  = offW2 + szW;
  size_t offList = offCnt + 256;
  size_t offGate = offList + (size_t)K_SLOT * E_NUM * T_TOK * sizeof(int);

  bf16*  Xb     = (bf16*)(ws + offXb);
  bf16*  W1b    = (bf16*)(ws + offW1);   // [E][F][H]
  bf16*  W2b    = (bf16*)(ws + offW2);   // [E][H][F]
  int*   counts = (int*)(ws + offCnt);
  int*   lists  = (int*)(ws + offList);
  float* gatesW = (float*)(ws + offGate);

  zero_counts_kernel<<<1, 32, 0, stream>>>(counts);
  route_kernel<<<T_TOK / 256, 256, 0, stream>>>(indices, probs, counts, lists, gatesW);

  cvt_x_kernel<<<(T_TOK * H_DIM / 4 + 255) / 256, 256, 0, stream>>>(hidden, Xb,
                                                                    T_TOK * H_DIM / 4);
  dim3 tb(32, 8);
  transpose_cvt_kernel<<<dim3(F_DIM / 32, H_DIM / 32, E_NUM), tb, 0, stream>>>(
      W1, W1b, H_DIM, F_DIM);   // [E][H][F] -> [E][F][H]
  transpose_cvt_kernel<<<dim3(H_DIM / 32, F_DIM / 32, E_NUM), tb, 0, stream>>>(
      W2, W2b, F_DIM, H_DIM);   // [E][F][H] -> [E][H][F]

  dim3 eg(T_TOK / MTILE, E_NUM);
  moe_expert_fused<<<eg, 256, 0, stream>>>(Xb, W1b, W2b, b1, b2, counts, lists,
                                           gatesW, out, /*pass=*/0, /*accum=*/0);
  moe_expert_fused<<<eg, 256, 0, stream>>>(Xb, W1b, W2b, b1, b2, counts, lists,
                                           gatesW, out, /*pass=*/1, /*accum=*/1);
}
